// CrossTaskInteractionLayer_4973572129019
// MI455X (gfx1250) — matrix operand order repaired and checked by reading, not compiled, not run on hardware
//
#include <hip/hip_runtime.h>
#include <hip/hip_bf16.h>

#define B_ 4096
#define T_ 8
#define D_ 2048
#define EPS_ 1e-5f
#define SCALE_INV 0.022097086912079608f  // 1/sqrt(2048)

typedef _Float16 v16h  __attribute__((ext_vector_type(16)));
typedef _Float16 half8 __attribute__((ext_vector_type(8)));
typedef _Float16 half4 __attribute__((ext_vector_type(4)));
typedef float    v8f   __attribute__((ext_vector_type(8)));

// ---------------------------------------------------------------------------
// Stage 1: fp32 -> fp16 conversion (vectorized x4)
// ---------------------------------------------------------------------------
__global__ void cvt_f32_f16_kernel(const float* __restrict__ in,
                                   _Float16* __restrict__ out, int n4) {
  int i = blockIdx.x * blockDim.x + threadIdx.x;
  if (i < n4) {
    float4 v = ((const float4*)in)[i];
    half4 h = { (_Float16)v.x, (_Float16)v.y, (_Float16)v.z, (_Float16)v.w };
    ((half4*)out)[i] = h;
  }
}

// ---------------------------------------------------------------------------
// Stage 2: projection GEMMs with v_wmma_f32_16x16x32_f16.
// O[b,t,e] = sum_d H[b,t,d] * W[e,d]   (W per-task for Q/G, shared for K/V)
// Each wave: 32 b-rows x 64 e-cols, fixed t (two A frags share four B frags
// -> 8 loads per 8 WMMAs per k-step). blockIdx.z = proj*8 + t.
// ---------------------------------------------------------------------------
__global__ __launch_bounds__(128) void proj_gemm_kernel(
    const _Float16* __restrict__ h16,
    const _Float16* __restrict__ wk16, const _Float16* __restrict__ wv16,
    const _Float16* __restrict__ wq16, const _Float16* __restrict__ wg16,
    _Float16* __restrict__ K16, _Float16* __restrict__ V16,
    _Float16* __restrict__ Q16, _Float16* __restrict__ G16) {
  const int wave = threadIdx.x >> 5;
  const int lane = threadIdx.x & 31;
  const int lr   = lane & 15;
  const int g    = lane >> 4;

  const int n0 = blockIdx.x * 64;
  const int b0 = blockIdx.y * 128 + wave * 32;
  const int z  = blockIdx.z;
  const int t  = z & 7;
  const int proj = z >> 3;

  const _Float16* W;
  _Float16* O;
  if (proj == 0)      { W = wk16;                               O = K16; }
  else if (proj == 1) { W = wv16;                               O = V16; }
  else if (proj == 2) { W = wq16 + (size_t)t * D_ * D_;         O = Q16; }
  else                { W = wg16 + (size_t)t * D_ * D_;         O = G16; }

  // A: rows (b0+lr, t) and (b0+16+lr, t) of H; two 8-half chunks each.
  const _Float16* aBase0 = h16 + ((size_t)(b0 + lr) * T_ + t) * D_;
  const _Float16* aBase1 = h16 + ((size_t)(b0 + 16 + lr) * T_ + t) * D_;
  // B: row (n0 + j*16 + lr) of W; one contiguous 16-half chunk per k-step.
  const _Float16* bBase0 = W + (size_t)(n0 +  0 + lr) * D_ + 16 * g;
  const _Float16* bBase1 = W + (size_t)(n0 + 16 + lr) * D_ + 16 * g;
  const _Float16* bBase2 = W + (size_t)(n0 + 32 + lr) * D_ + 16 * g;
  const _Float16* bBase3 = W + (size_t)(n0 + 48 + lr) * D_ + 16 * g;

  v8f c00 = {}, c01 = {}, c02 = {}, c03 = {};
  v8f c10 = {}, c11 = {}, c12 = {}, c13 = {};

  for (int k0 = 0; k0 < D_; k0 += 32) {
    half8 a0lo = *(const half8*)(aBase0 + k0 + 16 * g);
    half8 a0hi = *(const half8*)(aBase0 + k0 + 8 + 16 * g);
    half8 a1lo = *(const half8*)(aBase1 + k0 + 16 * g);
    half8 a1hi = *(const half8*)(aBase1 + k0 + 8 + 16 * g);
    v16h a0 = __builtin_shufflevector(a0lo, a0hi, 0, 1, 2, 3, 4, 5, 6, 7,
                                      8, 9, 10, 11, 12, 13, 14, 15);
    v16h a1 = __builtin_shufflevector(a1lo, a1hi, 0, 1, 2, 3, 4, 5, 6, 7,
                                      8, 9, 10, 11, 12, 13, 14, 15);
    v16h bf0 = *(const v16h*)(bBase0 + k0);
    v16h bf1 = *(const v16h*)(bBase1 + k0);
    v16h bf2 = *(const v16h*)(bBase2 + k0);
    v16h bf3 = *(const v16h*)(bBase3 + k0);
    c00 = __builtin_amdgcn_wmma_f32_16x16x32_f16(false, a0, false, bf0, (short)0, c00, false, false);
    c01 = __builtin_amdgcn_wmma_f32_16x16x32_f16(false, a0, false, bf1, (short)0, c01, false, false);
    c02 = __builtin_amdgcn_wmma_f32_16x16x32_f16(false, a0, false, bf2, (short)0, c02, false, false);
    c03 = __builtin_amdgcn_wmma_f32_16x16x32_f16(false, a0, false, bf3, (short)0, c03, false, false);
    c10 = __builtin_amdgcn_wmma_f32_16x16x32_f16(false, a1, false, bf0, (short)0, c10, false, false);
    c11 = __builtin_amdgcn_wmma_f32_16x16x32_f16(false, a1, false, bf1, (short)0, c11, false, false);
    c12 = __builtin_amdgcn_wmma_f32_16x16x32_f16(false, a1, false, bf2, (short)0, c12, false, false);
    c13 = __builtin_amdgcn_wmma_f32_16x16x32_f16(false, a1, false, bf3, (short)0, c13, false, false);
  }

  // Store: VGPR v of C holds (M = v + 8g, N = lr). Lanes 0-15 / 16-31 each
  // write 16 contiguous halves per row.
#pragma unroll
  for (int v = 0; v < 8; ++v) {
    const int row0 = b0 + v + 8 * g;
    const size_t ro0 = ((size_t)row0 * T_ + t) * D_ + n0 + lr;
    O[ro0 +  0] = (_Float16)c00[v];
    O[ro0 + 16] = (_Float16)c01[v];
    O[ro0 + 32] = (_Float16)c02[v];
    O[ro0 + 48] = (_Float16)c03[v];
    const int row1 = b0 + 16 + v + 8 * g;
    const size_t ro1 = ((size_t)row1 * T_ + t) * D_ + n0 + lr;
    O[ro1 +  0] = (_Float16)c10[v];
    O[ro1 + 16] = (_Float16)c11[v];
    O[ro1 + 32] = (_Float16)c12[v];
    O[ro1 + 48] = (_Float16)c13[v];
  }
}

// ---------------------------------------------------------------------------
// Stage 3: per-batch attention + gate + residual + LayerNorm.
// One block per b; wave w handles task t=w. K/V staged in LDS; x staged fp32.
// ---------------------------------------------------------------------------
__global__ __launch_bounds__(256) void attn_ln_kernel(
    const _Float16* __restrict__ K16, const _Float16* __restrict__ V16,
    const _Float16* __restrict__ Q16, const _Float16* __restrict__ G16,
    const float* __restrict__ H, const float* __restrict__ bg,
    const float* __restrict__ gamma, const float* __restrict__ beta,
    float* __restrict__ out) {
  __shared__ _Float16 sK[T_ * D_];   // 32 KB
  __shared__ _Float16 sV[T_ * D_];   // 32 KB
  __shared__ float    sX[T_ * D_];   // 64 KB

  const int b   = blockIdx.x;
  const int tid = threadIdx.x;
  const size_t base = (size_t)b * (T_ * D_);

  // Cooperative load of K,V tiles (uint4 = 8 halves per load).
  {
    const uint4* gk = (const uint4*)(K16 + base);
    const uint4* gv = (const uint4*)(V16 + base);
    uint4* lk = (uint4*)sK;
    uint4* lv = (uint4*)sV;
    for (int i = tid; i < (T_ * D_) / 8; i += 256) {
      lk[i] = gk[i];
      lv[i] = gv[i];
    }
  }
  __syncthreads();

  const int t    = tid >> 5;    // wave index == task index
  const int lane = tid & 31;

  // scores[t][s] = (1/sqrt(D)) * sum_e Q[b,t,e] * K[b,s,e]
  const _Float16* qrow = Q16 + base + (size_t)t * D_;
  float sc[T_];
#pragma unroll
  for (int s = 0; s < T_; ++s) sc[s] = 0.f;
  for (int i = 0; i < D_ / 32; ++i) {
    const int d = lane + 32 * i;
    const float q = (float)qrow[d];
#pragma unroll
    for (int s = 0; s < T_; ++s) sc[s] += q * (float)sK[s * D_ + d];
  }
#pragma unroll
  for (int s = 0; s < T_; ++s) {
    float v = sc[s];
    for (int off = 16; off > 0; off >>= 1) v += __shfl_xor(v, off, 32);
    sc[s] = v * SCALE_INV;
  }

  // softmax over s (replicated across lanes)
  float m = sc[0];
#pragma unroll
  for (int s = 1; s < T_; ++s) m = fmaxf(m, sc[s]);
  float den = 0.f;
#pragma unroll
  for (int s = 0; s < T_; ++s) { sc[s] = __expf(sc[s] - m); den += sc[s]; }
  const float rden = 1.f / den;
#pragma unroll
  for (int s = 0; s < T_; ++s) sc[s] *= rden;

  // ctx + sigmoid gate + residual; accumulate LN statistics
  const _Float16* grow = G16 + base + (size_t)t * D_;
  const float* hrow = H + base + (size_t)t * D_;
  float sum = 0.f, sumsq = 0.f;
  for (int i = 0; i < D_ / 32; ++i) {
    const int d = lane + 32 * i;
    float c = 0.f;
#pragma unroll
    for (int s = 0; s < T_; ++s) c += sc[s] * (float)sV[s * D_ + d];
    const float gpre = (float)grow[d] + bg[t * D_ + d];
    const float gate = 1.f / (1.f + __expf(-gpre));
    const float x = hrow[d] + gate * c;
    sX[t * D_ + d] = x;
    sum += x;
    sumsq += x * x;
  }
  for (int off = 16; off > 0; off >>= 1) {
    sum   += __shfl_xor(sum, off, 32);
    sumsq += __shfl_xor(sumsq, off, 32);
  }
  const float mu   = sum * (1.f / D_);
  const float var  = sumsq * (1.f / D_) - mu * mu;
  const float rstd = rsqrtf(var + EPS_);

  float* orow = out + base + (size_t)t * D_;
  for (int i = 0; i < D_ / 32; ++i) {
    const int d = lane + 32 * i;
    const float x = sX[t * D_ + d];
    orow[d] = (x - mu) * rstd * gamma[t * D_ + d] + beta[t * D_ + d];
  }
}

// ---------------------------------------------------------------------------
extern "C" void kernel_launch(void* const* d_in, const int* in_sizes, int n_in,
                              void* d_out, int out_size, void* d_ws, size_t ws_size,
                              hipStream_t stream) {
  (void)in_sizes; (void)n_in; (void)out_size; (void)ws_size;

  const float* H     = (const float*)d_in[0];  // (B,T,D)
  const float* Wq    = (const float*)d_in[1];  // (T,D,D)
  const float* Wk    = (const float*)d_in[2];  // (D,D)
  const float* Wv    = (const float*)d_in[3];  // (D,D)
  const float* Wg    = (const float*)d_in[4];  // (T,D,D)
  const float* bg    = (const float*)d_in[5];  // (T,D)
  const float* gamma = (const float*)d_in[6];  // (T,D)
  const float* beta  = (const float*)d_in[7];  // (T,D)
  float* out = (float*)d_out;

  const size_t HT = (size_t)B_ * T_ * D_;      // 67,108,864
  const size_t DD = (size_t)D_ * D_;           //  4,194,304

  char* ws = (char*)d_ws;
  _Float16* h16 = (_Float16*)(ws);                              size_t off = HT * 2;
  _Float16* wq16 = (_Float16*)(ws + off);  off += T_ * DD * 2;
  _Float16* wk16 = (_Float16*)(ws + off);  off += DD * 2;
  _Float16* wv16 = (_Float16*)(ws + off);  off += DD * 2;
  _Float16* wg16 = (_Float16*)(ws + off);  off += T_ * DD * 2;
  _Float16* K16 = (_Float16*)(ws + off);   off += HT * 2;
  _Float16* V16 = (_Float16*)(ws + off);   off += HT * 2;
  _Float16* Q16 = (_Float16*)(ws + off);   off += HT * 2;
  _Float16* G16 = (_Float16*)(ws + off);   off += HT * 2;

  // Stage 1: conversions
  {
    const int blk = 256;
    int n4;
    n4 = (int)(HT / 4);
    cvt_f32_f16_kernel<<<(n4 + blk - 1) / blk, blk, 0, stream>>>(H, h16, n4);
    n4 = (int)(T_ * DD / 4);
    cvt_f32_f16_kernel<<<(n4 + blk - 1) / blk, blk, 0, stream>>>(Wq, wq16, n4);
    n4 = (int)(DD / 4);
    cvt_f32_f16_kernel<<<(n4 + blk - 1) / blk, blk, 0, stream>>>(Wk, wk16, n4);
    cvt_f32_f16_kernel<<<(n4 + blk - 1) / blk, blk, 0, stream>>>(Wv, wv16, n4);
    n4 = (int)(T_ * DD / 4);
    cvt_f32_f16_kernel<<<(n4 + blk - 1) / blk, blk, 0, stream>>>(Wg, wg16, n4);
  }

  // Stage 2: 4 projection GEMMs (K,V shared weight; Q,G per-task)
  {
    dim3 grid(D_ / 64, B_ / 128, 4 * T_);  // (32, 32, 32)
    proj_gemm_kernel<<<grid, 128, 0, stream>>>(h16, wk16, wv16, wq16, wg16,
                                               K16, V16, Q16, G16);
  }

  // Stage 3: attention + gate + residual + LayerNorm
  attn_ln_kernel<<<B_, 256, 0, stream>>>(K16, V16, Q16, G16, H, bg, gamma,
                                         beta, out);
}
